// MHA_429496729616
// MI455X (gfx1250) — matrix operand order repaired and checked by reading, not compiled, not run on hardware
//
#include <hip/hip_runtime.h>

// ---------------------------------------------------------------------------
// MI455X (gfx1250) causal MHA prefill.
// One-time fp32->bf16 convert/transpose pre-passes, then all hot loops use
// double-buffered GLOBAL_LOAD_ASYNC_TO_LDS_B128 staging (ASYNCcnt) feeding
// V_WMMA_F32_16X16X32_BF16 (fp32 accumulate).
// ---------------------------------------------------------------------------

typedef __attribute__((ext_vector_type(16))) __bf16 bf16x16;
typedef __attribute__((ext_vector_type(8)))  __bf16 bf16x8;
typedef __attribute__((ext_vector_type(4)))  __bf16 bf16x4;
typedef __attribute__((ext_vector_type(8)))  float  f32x8;

static __device__ __forceinline__ f32x8 zero8() {
  f32x8 z;
#pragma unroll
  for (int i = 0; i < 8; ++i) z[i] = 0.0f;
  return z;
}

static __device__ __forceinline__ bf16x16 join16(bf16x8 lo, bf16x8 hi) {
  bf16x16 r;
#pragma unroll
  for (int i = 0; i < 8; ++i) { r[i] = lo[i]; r[i + 8] = hi[i]; }
  return r;
}

// A-operand fragment (16-bit 16x32): 16 contiguous K values;
// caller passes base = row*pitch + 16*(lane>>4).
static __device__ __forceinline__ bf16x16 lds_frag_a(const __bf16* p) {
  return join16(*(const bf16x8*)(p), *(const bf16x8*)(p + 8));
}
// B-operand fragment (16-bit 32x16): 16 contiguous K values;
// caller passes base = col*pitch + 16*(lane>>4).
static __device__ __forceinline__ bf16x16 lds_frag_b(const __bf16* p) {
  return join16(*(const bf16x8*)(p), *(const bf16x8*)(p + 8));
}

// ---- CDNA5 async copy: global -> LDS, 16B per lane, tracked by ASYNCcnt ----
static __device__ __forceinline__ void async_b128(const void* g, void* l) {
  unsigned lds_off = (unsigned)(size_t)l;             // low 32b of generic = LDS offset
  unsigned long long ga = (unsigned long long)(size_t)g;
  asm volatile("global_load_async_to_lds_b128 %0, %1, off"
               :: "v"(lds_off), "v"(ga) : "memory");
}
static __device__ __forceinline__ void wait_async_le4() {
  asm volatile("s_wait_asynccnt 0x4" ::: "memory");
}
static __device__ __forceinline__ void wait_async_0() {
  asm volatile("s_wait_asynccnt 0x0" ::: "memory");
}

// ---------------------------------------------------------------------------
// Pre-pass 1: elementwise fp32 -> bf16 (vectorized float4 -> bf16x4)
// ---------------------------------------------------------------------------
__global__ __launch_bounds__(256) void cvt_bf16(const float* __restrict__ in,
                                                __bf16* __restrict__ out) {
  const int idx = blockIdx.x * 256 + threadIdx.x;
  float4 f = *(const float4*)(in + (size_t)idx * 4);
  bf16x4 h;
  h[0] = (__bf16)f.x; h[1] = (__bf16)f.y; h[2] = (__bf16)f.z; h[3] = (__bf16)f.w;
  *(bf16x4*)(out + (size_t)idx * 4) = h;
}

// ---------------------------------------------------------------------------
// Pre-pass 2: W[K,N] fp32 -> Wt[N,K] bf16  (LDS-tiled 32x32 transpose)
// ---------------------------------------------------------------------------
__global__ __launch_bounds__(256) void transpose_cvt(
    const float* __restrict__ W, __bf16* __restrict__ Wt, int K, int N) {
  __shared__ __bf16 Tl[32 * 33];
  const int tid = threadIdx.x;
  const int Nb = blockIdx.x * 32, Kb = blockIdx.y * 32;
#pragma unroll
  for (int u = 0; u < 4; ++u) {
    const int idx = tid + u * 256;
    const int kk = idx >> 5, nn = idx & 31;
    Tl[nn * 33 + kk] = (__bf16)W[(size_t)(Kb + kk) * N + Nb + nn];
  }
  __syncthreads();
#pragma unroll
  for (int u = 0; u < 4; ++u) {
    const int idx = tid + u * 256;
    const int nn = idx >> 5, kk = idx & 31;
    Wt[(size_t)(Nb + nn) * K + Kb + kk] = Tl[nn * 33 + kk];
  }
}

// ---------------------------------------------------------------------------
// Tiled GEMM: C[M,N] = A[M,K](bf16) @ Wt[N,K](bf16)^T + bias[N]
// Block 256 thr (8 waves), tile 128x128, K-step 32, double-buffered async LDS.
// MODE 0: fp32 store to outF (stride N).
// MODE 1: QKV epilogue: q -> qb (bf16, pre-scaled 1/8, [B,h,T,hd])
//                       k -> koutF (fp32 output) + kbB (bf16, [B,h,T,hd])
//                       v -> voutF (fp32 output) + vtB (bf16, [B,h,hd,T])
// ---------------------------------------------------------------------------
template <int MODE>
__global__ __launch_bounds__(256) void gemm_wmma(
    const __bf16* __restrict__ A, const __bf16* __restrict__ Wt,
    const float* __restrict__ bias, int K, int N,
    float* __restrict__ outF, float* __restrict__ koutF, float* __restrict__ voutF,
    __bf16* __restrict__ qbB, __bf16* __restrict__ kbB, __bf16* __restrict__ vtB)
{
  __shared__ __align__(16) __bf16 Al[2][128 * 40];  // [m][k], pitch 40 (80B)
  __shared__ __align__(16) __bf16 Bl[2][128 * 40];  // [n][k], pitch 40 (80B)

  const int tid  = threadIdx.x;
  const int lane = tid & 31;
  const int wave = tid >> 5;
  const int wm   = wave & 1;           // 2 waves along M (64 rows each)
  const int wn   = wave >> 1;          // 4 waves along N (32 cols each)
  const int half = lane >> 4;
  const int l16  = lane & 15;
  const int Mbase = blockIdx.y * 128;
  const int Nbase = blockIdx.x * 128;

  f32x8 acc[4][2];
#pragma unroll
  for (int i = 0; i < 4; ++i)
#pragma unroll
    for (int j = 0; j < 2; ++j) acc[i][j] = zero8();

  // issue one 128x32 A tile + 128x32 B tile (4 async b128 per thread)
  auto issue_tile = [&](int buf, int k0) {
#pragma unroll
    for (int u = 0; u < 2; ++u) {
      const int idx = tid + u * 256;
      const int m = idx >> 2, c = (idx & 3) * 8;   // 8 bf16 = 16B chunk
      async_b128(A + (size_t)(Mbase + m) * K + k0 + c, &Al[buf][m * 40 + c]);
    }
#pragma unroll
    for (int u = 0; u < 2; ++u) {
      const int idx = tid + u * 256;
      const int n = idx >> 2, c = (idx & 3) * 8;
      async_b128(Wt + (size_t)(Nbase + n) * K + k0 + c, &Bl[buf][n * 40 + c]);
    }
  };

  const int nk = K / 32;
  issue_tile(0, 0);
  for (int it = 0; it < nk; ++it) {
    const int cur = it & 1;
    if (it + 1 < nk) { issue_tile(cur ^ 1, (it + 1) * 32); wait_async_le4(); }
    else             { wait_async_0(); }
    __syncthreads();

    bf16x16 af[4];
#pragma unroll
    for (int i = 0; i < 4; ++i)
      af[i] = lds_frag_a(&Al[cur][(wm * 64 + i * 16 + l16) * 40 + half * 16]);
    bf16x16 bfr[2];
#pragma unroll
    for (int j = 0; j < 2; ++j)
      bfr[j] = lds_frag_b(&Bl[cur][(wn * 32 + j * 16 + l16) * 40 + half * 16]);

#pragma unroll
    for (int i = 0; i < 4; ++i)
#pragma unroll
      for (int j = 0; j < 2; ++j)
        acc[i][j] = __builtin_amdgcn_wmma_f32_16x16x32_bf16(
            false, af[i], false, bfr[j], (short)0, acc[i][j], false, false);
    __syncthreads();
  }

  // ---- epilogue. C layout: row = r + 8*half, col = l16 ----
#pragma unroll
  for (int i = 0; i < 4; ++i) {
#pragma unroll
    for (int j = 0; j < 2; ++j) {
      const int col = Nbase + wn * 32 + j * 16 + l16;
      const float bia = bias[col];
#pragma unroll
      for (int r = 0; r < 8; ++r) {
        const int row = Mbase + wm * 64 + i * 16 + half * 8 + r;
        const float val = acc[i][j][r] + bia;
        if (MODE == 0) {
          outF[(size_t)row * N + col] = val;
        } else {
          const int sel = col >> 10;            // 0:q 1:k 2:v
          const int cc  = col & 1023;
          const int head = cc >> 6, hd = cc & 63;
          const int b = row >> 10, t = row & 1023;
          const int bh = (b << 4) + head;
          const size_t o = ((size_t)bh * 1024 + t) * 64 + hd;
          if (sel == 0) {
            qbB[o] = (__bf16)(val * 0.125f);                  // pre-scaled 1/sqrt(64)
          } else if (sel == 1) {
            koutF[o] = val; kbB[o] = (__bf16)val;
          } else {
            voutF[o] = val;
            vtB[((size_t)bh * 64 + hd) * 1024 + t] = (__bf16)val;  // transposed
          }
        }
      }
    }
  }
}

// ---------------------------------------------------------------------------
// Flash-style causal attention. Block = 128 thr (4 waves), one (b,h), 64 q
// rows (wave w owns rows qt*64+w*16..+15). 32-key tiles, double-buffered
// async staging of K [key][hd] and V^T [hd][key] (already bf16 in ws).
// Writes y as bf16 [B,T,D] for the projection GEMM.
// ---------------------------------------------------------------------------
__global__ __launch_bounds__(128) void attn_wmma(
    const __bf16* __restrict__ q, const __bf16* __restrict__ k,
    const __bf16* __restrict__ vt, __bf16* __restrict__ y)
{
  __shared__ __align__(16) __bf16 Kl[2][32 * 72];   // [key][hd], pitch 72
  __shared__ __align__(16) __bf16 Vl[2][64 * 40];   // [hd][key], pitch 40
  __shared__ __align__(16) __bf16 Pl[4 * 16 * 40];  // per-wave P, pitch 40

  const int tid  = threadIdx.x;
  const int lane = tid & 31;
  const int wave = tid >> 5;
  const int half = lane >> 4;
  const int l16  = lane & 15;
  const int bh = blockIdx.y;                 // b*16 + head
  const int qt = blockIdx.x;                 // 64-row query tile
  const int qbase = qt * 64 + wave * 16;
  const size_t bhbase = (size_t)bh * 1024 * 64;

  // Q fragments straight from global bf16 (already scaled by 1/8)
  bf16x16 qf[2];
#pragma unroll
  for (int g = 0; g < 2; ++g) {
    const __bf16* qp = q + bhbase + (size_t)(qbase + l16) * 64 + g * 32 + half * 16;
    qf[g] = join16(*(const bf16x8*)qp, *(const bf16x8*)(qp + 8));
  }

  float m_old[8], l_sum[8];
#pragma unroll
  for (int r = 0; r < 8; ++r) { m_old[r] = -1e30f; l_sum[r] = 0.0f; }
  f32x8 yacc[4];
#pragma unroll
  for (int c = 0; c < 4; ++c) yacc[c] = zero8();

  // stage one 32-key tile: K 32x128B + V^T 64x64B  (4 async b128 per thread)
  auto issue_tile = [&](int buf, int kb0) {
#pragma unroll
    for (int u = 0; u < 2; ++u) {
      const int idx = tid + u * 128;
      const int key = idx >> 3, c = (idx & 7) * 8;
      async_b128(k + bhbase + (size_t)(kb0 + key) * 64 + c, &Kl[buf][key * 72 + c]);
    }
#pragma unroll
    for (int u = 0; u < 2; ++u) {
      const int idx = tid + u * 128;
      const int hd = idx >> 2, c = (idx & 3) * 8;
      async_b128(vt + (size_t)bh * 64 * 1024 + (size_t)hd * 1024 + kb0 + c,
                 &Vl[buf][hd * 40 + c]);
    }
  };

  const int ntiles = qt * 2 + 2;             // causal: keys <= qt*64+63
  issue_tile(0, 0);
  for (int kt = 0; kt < ntiles; ++kt) {
    const int cur = kt & 1;
    const int kb0 = kt * 32;
    if (kt + 1 < ntiles) { issue_tile(cur ^ 1, kb0 + 32); wait_async_le4(); }
    else                 { wait_async_0(); }
    __syncthreads();

    if (kb0 <= qbase + 15) {                 // wave-uniform (EXEC all-1 for WMMA)
      // ---- scores S = Q K^T : 16x32, 4 WMMAs ----
      f32x8 s[2];
#pragma unroll
      for (int j = 0; j < 2; ++j) {
        s[j] = zero8();
        bf16x16 kf0 = lds_frag_b(&Kl[cur][(j * 16 + l16) * 72 + half * 16]);
        bf16x16 kf1 = lds_frag_b(&Kl[cur][(j * 16 + l16) * 72 + 32 + half * 16]);
        s[j] = __builtin_amdgcn_wmma_f32_16x16x32_bf16(
            false, qf[0], false, kf0, (short)0, s[j], false, false);
        s[j] = __builtin_amdgcn_wmma_f32_16x16x32_bf16(
            false, qf[1], false, kf1, (short)0, s[j], false, false);
      }
      // ---- causal mask on diagonal tiles ----
      if (kb0 + 31 > qbase) {
#pragma unroll
        for (int j = 0; j < 2; ++j) {
          const int key = kb0 + j * 16 + l16;
#pragma unroll
          for (int r = 0; r < 8; ++r)
            if (key > qbase + half * 8 + r) s[j][r] = -1e30f;
        }
      }
      // ---- online softmax (row spans 16 lanes of one half-wave) ----
      float tmax[8], psum[8];
#pragma unroll
      for (int r = 0; r < 8; ++r) tmax[r] = fmaxf(s[0][r], s[1][r]);
#pragma unroll
      for (int m = 1; m < 16; m <<= 1)
#pragma unroll
        for (int r = 0; r < 8; ++r)
          tmax[r] = fmaxf(tmax[r], __shfl_xor(tmax[r], m, 32));
#pragma unroll
      for (int r = 0; r < 8; ++r) {
        const float mn = fmaxf(m_old[r], tmax[r]);
        const float sc = __expf(m_old[r] - mn);
        m_old[r] = mn;
        const float p0 = __expf(s[0][r] - mn);
        const float p1 = __expf(s[1][r] - mn);
        s[0][r] = p0; s[1][r] = p1;
        psum[r] = p0 + p1;
        l_sum[r] *= sc;
#pragma unroll
        for (int c = 0; c < 4; ++c) yacc[c][r] *= sc;
      }
#pragma unroll
      for (int m = 1; m < 16; m <<= 1)
#pragma unroll
        for (int r = 0; r < 8; ++r) psum[r] += __shfl_xor(psum[r], m, 32);
#pragma unroll
      for (int r = 0; r < 8; ++r) l_sum[r] += psum[r];

      // ---- P via wave-private LDS (C-layout -> A-layout) ----
      __bf16* pw = &Pl[wave * 16 * 40];
#pragma unroll
      for (int r = 0; r < 8; ++r) {
        const int row = half * 8 + r;
        pw[row * 40 + l16]      = (__bf16)s[0][r];
        pw[row * 40 + 16 + l16] = (__bf16)s[1][r];
      }
      asm volatile("s_wait_dscnt 0x0" ::: "memory");   // wave-local RAW on LDS
      bf16x16 pf = lds_frag_a(&pw[l16 * 40 + half * 16]);

      // ---- yacc += P @ V : 4 WMMAs ----
#pragma unroll
      for (int c = 0; c < 4; ++c) {
        bf16x16 vf = lds_frag_b(&Vl[cur][(c * 16 + l16) * 40 + half * 16]);
        yacc[c] = __builtin_amdgcn_wmma_f32_16x16x32_bf16(
            false, pf, false, vf, (short)0, yacc[c], false, false);
      }
    }
    __syncthreads();
  }

  // ---- normalize + store y (bf16, [B,T,D]) ----
  const int b = bh >> 4, head = bh & 15;
#pragma unroll
  for (int c = 0; c < 4; ++c)
#pragma unroll
    for (int r = 0; r < 8; ++r) {
      const int t = qbase + half * 8 + r;
      const int col = head * 64 + c * 16 + l16;
      y[((size_t)(b * 1024 + t)) * 1024 + col] = (__bf16)(yacc[c][r] / l_sum[r]);
    }
}

// ---------------------------------------------------------------------------
extern "C" void kernel_launch(void* const* d_in, const int* in_sizes, int n_in,
                              void* d_out, int out_size, void* d_ws, size_t ws_size,
                              hipStream_t stream) {
  const float* x     = (const float*)d_in[0];   // [8,1024,1024]
  const float* Wqkv  = (const float*)d_in[1];   // [1024,3072]
  const float* bqkv  = (const float*)d_in[2];   // [3072]
  const float* Wproj = (const float*)d_in[3];   // [1024,1024]
  const float* bproj = (const float*)d_in[4];   // [1024]

  const size_t M8 = (size_t)8 * 1024 * 1024;    // 8M elements
  float* outp = (float*)d_out;
  float* kout = outp + M8;                      // fp32 [B,h,T,hd] (output)
  float* vout = outp + 2 * M8;                  // fp32 [B,h,T,hd] (output)

  __bf16* xb     = (__bf16*)d_ws;               // 8M  [8192,1024]
  __bf16* wqkvt  = xb + M8;                     // 3M  [3072,1024] (W^T)
  __bf16* wprojt = wqkvt + (size_t)3 * 1024 * 1024;  // 1M [1024,1024] (W^T)
  __bf16* qb     = wprojt + (size_t)1024 * 1024;     // 8M [B,h,T,hd], pre-scaled
  __bf16* kbb    = qb + M8;                     // 8M [B,h,T,hd]
  __bf16* vtb    = kbb + M8;                    // 8M [B,h,hd,T]
  __bf16* yb     = vtb + M8;                    // 8M [B,T,D]

  // 0) one-time convert / transpose pre-passes
  cvt_bf16<<<8192, 256, 0, stream>>>(x, xb);
  transpose_cvt<<<dim3(3072 / 32, 1024 / 32), 256, 0, stream>>>(Wqkv, wqkvt, 1024, 3072);
  transpose_cvt<<<dim3(1024 / 32, 1024 / 32), 256, 0, stream>>>(Wproj, wprojt, 1024, 1024);

  // 1) QKV GEMM: [8192,1024]x[1024,3072] + bias, scatter q/k/v
  gemm_wmma<1><<<dim3(3072 / 128, 8192 / 128), 256, 0, stream>>>(
      xb, wqkvt, bqkv, 1024, 3072, nullptr, kout, vout, qb, kbb, vtb);

  // 2) causal flash attention: 128 (b,h) x 16 q-tiles
  attn_wmma<<<dim3(1024 / 64, 8 * 16), 128, 0, stream>>>(qb, kbb, vtb, yb);

  // 3) output projection: [8192,1024]x[1024,1024] + bias
  gemm_wmma<0><<<dim3(1024 / 128, 8192 / 128), 256, 0, stream>>>(
      yb, wprojt, bproj, 1024, 1024, outp, nullptr, nullptr, nullptr, nullptr, nullptr);
}
